// Target_Lstm_2138893713849
// MI455X (gfx1250) — matrix operand order repaired and checked by reading, not compiled, not run on hardware
//
#include <hip/hip_runtime.h>
#include <hip/hip_bf16.h>

#define VOCAB 32000
#define EMB   200
#define HID   128
#define BATCH 64
#define TSEQ  64
#define G4    (4*HID)   // 512 gate columns

typedef __attribute__((ext_vector_type(16))) __bf16        vbf16;
typedef __attribute__((ext_vector_type(8)))  float         v8f;
typedef __attribute__((ext_vector_type(4)))  unsigned int  v4u;
typedef __attribute__((ext_vector_type(8)))  int           v8i;
typedef __attribute__((ext_vector_type(4)))  int           v4i;

#if defined(__has_builtin)
#if __has_builtin(__builtin_amdgcn_tensor_load_to_lds)
#define HAVE_TDM 1
#endif
#endif

__device__ __forceinline__ unsigned short f2bf(float f) {
    unsigned int u = __float_as_uint(f);
    unsigned int r = u + 0x7fffu + ((u >> 16) & 1u);   // round-to-nearest-even
    return (unsigned short)(r >> 16);
}
__device__ __forceinline__ float sigm(float x) { return 1.0f / (1.0f + __expf(-x)); }

// ---------------------------------------------------------------------------
// Phase 1: sequential 2-layer LSTM. One workgroup per batch row (recurrence is
// serial in T but independent across B). 512 threads = one per gate column.
// Emits h1 history as bf16 rows Hb[(t*BATCH+b)][HID] for the phase-2 GEMM.
// ---------------------------------------------------------------------------
__global__ __launch_bounds__(512)
void lstm_seq(const int* __restrict__ tokens, const float* __restrict__ emb,
              const float* __restrict__ W0, const float* __restrict__ U0,
              const float* __restrict__ b0,
              const float* __restrict__ W1, const float* __restrict__ U1,
              const float* __restrict__ b1,
              unsigned short* __restrict__ Hb)
{
    __shared__ float x[EMB];
    __shared__ float h0[HID], c0[HID], h1[HID], c1[HID];
    __shared__ float z[G4];

    const int b = blockIdx.x;
    const int j = threadIdx.x;           // 0..511

    if (j < HID) { h0[j] = 0.f; c0[j] = 0.f; h1[j] = 0.f; c1[j] = 0.f; }
    __syncthreads();

    for (int t = 0; t < TSEQ; ++t) {
        const int tok = tokens[b * TSEQ + t];
        for (int k = j; k < EMB; k += 512) x[k] = emb[(size_t)tok * EMB + k];
        __syncthreads();

        // layer 0: z = x@W0 + h0@U0 + b0   (coalesced weight rows, W0[k][512])
        float acc = b0[j];
        for (int k = 0; k < EMB; ++k) acc = fmaf(x[k],  W0[k * G4 + j], acc);
        for (int k = 0; k < HID; ++k) acc = fmaf(h0[k], U0[k * G4 + j], acc);
        z[j] = acc;
        __syncthreads();

        if (j < HID) {   // gate order i,f,g,o
            float i_ = sigm(z[j]);
            float f_ = sigm(z[HID + j]);
            float g_ = tanhf(z[2 * HID + j]);
            float o_ = sigm(z[3 * HID + j]);
            float c  = f_ * c0[j] + i_ * g_;
            c0[j] = c;
            h0[j] = o_ * tanhf(c);
        }
        __syncthreads();

        // layer 1: z = h0@W1 + h1@U1 + b1
        acc = b1[j];
        for (int k = 0; k < HID; ++k) acc = fmaf(h0[k], W1[k * G4 + j], acc);
        for (int k = 0; k < HID; ++k) acc = fmaf(h1[k], U1[k * G4 + j], acc);
        z[j] = acc;
        __syncthreads();

        if (j < HID) {
            float i_ = sigm(z[j]);
            float f_ = sigm(z[HID + j]);
            float g_ = tanhf(z[2 * HID + j]);
            float o_ = sigm(z[3 * HID + j]);
            float c  = f_ * c1[j] + i_ * g_;
            float h  = o_ * tanhf(c);
            c1[j] = c;
            h1[j] = h;
            Hb[((size_t)t * BATCH + b) * HID + j] = f2bf(h);   // bf16 GEMM operand
        }
        __syncthreads();
    }
}

// ---------------------------------------------------------------------------
// Phase 2a: transpose + convert Wout[K=128][N=32000] f32 -> WoutT[N][K] bf16.
// ---------------------------------------------------------------------------
__global__ __launch_bounds__(256)
void wout_transpose(const float* __restrict__ Wout, unsigned short* __restrict__ WoutT)
{
    __shared__ float tile[32][33];
    const int n0 = blockIdx.x * 32;
    const int k0 = blockIdx.y * 32;
    const int tx = threadIdx.x & 31;
    const int ty = threadIdx.x >> 5;

    for (int r = ty; r < 32; r += 8)
        tile[r][tx] = Wout[(size_t)(k0 + r) * VOCAB + n0 + tx];
    __syncthreads();
    for (int r = ty; r < 32; r += 8)
        WoutT[(size_t)(n0 + r) * HID + k0 + tx] = f2bf(tile[tx][r]);
}

// ---------------------------------------------------------------------------
// Phase 2b: out = relu(H @ Wout + bout), M=4096 N=32000 K=128, bf16 WMMA,
// f32 accumulate. Block = 8 waves; all waves share one 64-column B slab of
// WoutT staged in LDS via the Tensor Data Mover (8x L2-traffic reduction);
// each wave owns 1 M-tile x 4 N-tiles, K unrolled 4x32.
//   A frag (16-bit A 16x32): lane m=lane%16; pair K = (v>=4?16:0)+(hi?8:0)+2*(v&3)
//   B frag (16-bit B 32x16): lane n=lane%16; K = (v>=4?16:0)+(hi?8:0)+2*(v&3)
//   -> two 8-half runs -> two ds_load_b128 from the LDS slab.
// ---------------------------------------------------------------------------
__global__ __launch_bounds__(256)
void out_gemm(const unsigned short* __restrict__ Hb,
              const unsigned short* __restrict__ WoutT,
              const float* __restrict__ bout, float* __restrict__ out)
{
    __shared__ unsigned short Bsh[64 * HID];     // 16 KB: rows n0..n0+63 of WoutT

    const int lane  = threadIdx.x & 31;
    const int wave  = threadIdx.x >> 5;          // 0..7
    const int mtile = blockIdx.y * 8 + wave;     // 0..255
    const int n0    = blockIdx.x * 64;           // 4 consecutive N-tiles
    const int ln    = lane & 15;
    const int hi    = lane >> 4;                 // 0/1 half-wave

#ifdef HAVE_TDM
    // ---- Tensor Data Mover: 2-D tile WoutT[n0..n0+63][0..127] -> LDS ------
    if (wave == 0) {
        const unsigned long long ga =
            (unsigned long long)(const void*)(WoutT + (size_t)n0 * HID);
        v4u g0 = {0, 0, 0, 0};
        g0.x = 1u;                                   // count=1 (valid descriptor)
        g0.y = 0u;                                   // lds_addr = 0 (Bsh is first/only LDS)
        g0.z = (unsigned int)(ga & 0xffffffffu);     // global_addr[31:0]
        g0.w = (unsigned int)((ga >> 32) & 0x01ffffffu) | (2u << 30); // addr[56:32] | type=2
        union { v8i v; unsigned int u[8]; } g1;
        g1.u[0] = (1u << 16);                        // wg_mask=0, data_size=1 (2 bytes)
        g1.u[1] = ((unsigned)HID & 0xffffu) << 16;   // tensor_dim0[15:0] at bits[63:48]
        g1.u[2] = ((unsigned)HID >> 16) | ((unsigned)(VOCAB & 0xffff) << 16); // dim0 hi | dim1 lo
        g1.u[3] = ((unsigned)VOCAB >> 16) | ((unsigned)HID << 16);  // dim1 hi | tile_dim0=128
        g1.u[4] = 64u;                               // tile_dim1=64, tile_dim2=0
        g1.u[5] = (unsigned)HID;                     // tensor_dim0_stride = 128 (low 32)
        g1.u[6] = 0u;
        g1.u[7] = 0u;
        v4i g2 = {0, 0, 0, 0};                       // 2-D tensor: groups 2/3 unused
        v4i g3 = {0, 0, 0, 0};
#if __clang_major__ >= 23
        v8i g4 = {0, 0, 0, 0, 0, 0, 0, 0};
        __builtin_amdgcn_tensor_load_to_lds(g0, g1.v, g2, g3, g4, 0);
#else
        __builtin_amdgcn_tensor_load_to_lds(g0, g1.v, g2, g3, 0);
#endif
        __builtin_amdgcn_s_wait_tensorcnt(0);
    }
    // The TDM builtin is not modeled as an LDS store, so without a visible
    // store LLVM folds all Bsh loads to undef. This opaque, never-taken
    // kernarg-guarded store keeps the slab "written" without runtime effect.
    if ((unsigned long long)(const void*)out == 0ull) Bsh[threadIdx.x] = 1;
#else
    // ---- fallback: cooperative global -> LDS copy (16 B per thread-iter) --
    {
        const uint4* src = (const uint4*)(WoutT + (size_t)n0 * HID);
        uint4*       dst = (uint4*)Bsh;
        for (int i = threadIdx.x; i < 64 * HID / 8; i += 256) dst[i] = src[i];
    }
#endif
    __syncthreads();   // LDS slab visible to all 8 waves

    v8f acc[4] = {v8f{}, v8f{}, v8f{}, v8f{}};
    const unsigned short* arow = Hb + (size_t)(mtile * 16 + ln) * HID;

    #pragma unroll
    for (int kk = 0; kk < HID; kk += 32) {
        union { vbf16 v; unsigned int u[8]; } A;
        #pragma unroll
        for (int v = 0; v < 8; ++v) {
            const int K = kk + ((v >= 4) ? 16 : 0) + (hi ? 8 : 0) + 2 * (v & 3);
            A.u[v] = *(const unsigned int*)(arow + K);
        }
        #pragma unroll
        for (int t = 0; t < 4; ++t) {
            const unsigned short* bsrc = Bsh + (t * 16 + ln) * HID + kk + hi * 8;
            union { vbf16 v; uint4 q[2]; } B;
            B.q[0] = *(const uint4*)(bsrc);         // K 8h..8h+7        (ds_load_b128)
            B.q[1] = *(const uint4*)(bsrc + 16);    // K 16+8h..16+8h+7  (ds_load_b128)
            acc[t] = __builtin_amdgcn_wmma_f32_16x16x32_bf16(
                false, A.v, false, B.v, (short)0, acc[t], false, false);
        }
    }

    // fused bias + ReLU store; remap GEMM row R = t*64+b -> out[b][t][:]
    #pragma unroll
    for (int t = 0; t < 4; ++t) {
        const int col  = n0 + t * 16 + ln;
        const float bias = bout[col];
        #pragma unroll
        for (int r = 0; r < 8; ++r) {
            const int R  = mtile * 16 + r + hi * 8;
            const int bb = R & (BATCH - 1);
            const int ts = R >> 6;
            float v = acc[t][r] + bias;
            out[((size_t)bb * TSEQ + ts) * VOCAB + col] = v > 0.f ? v : 0.f;
        }
    }
}

// ---------------------------------------------------------------------------
extern "C" void kernel_launch(void* const* d_in, const int* in_sizes, int n_in,
                              void* d_out, int out_size, void* d_ws, size_t ws_size,
                              hipStream_t stream)
{
    const int*   tokens = (const int*)  d_in[0];
    const float* emb    = (const float*)d_in[1];
    const float* W0     = (const float*)d_in[2];
    const float* U0     = (const float*)d_in[3];
    const float* b0     = (const float*)d_in[4];
    const float* W1     = (const float*)d_in[5];
    const float* U1     = (const float*)d_in[6];
    const float* b1     = (const float*)d_in[7];
    const float* Wout   = (const float*)d_in[8];
    const float* bout   = (const float*)d_in[9];
    float*       out    = (float*)d_out;

    unsigned short* Hb    = (unsigned short*)d_ws;                      // 4096*128*2 = 1 MB
    unsigned short* WoutT = (unsigned short*)((char*)d_ws + (1 << 20)); // 32000*128*2 = 8 MB

    lstm_seq<<<BATCH, 512, 0, stream>>>(tokens, emb, W0, U0, b0, W1, U1, b1, Hb);
    wout_transpose<<<dim3(VOCAB / 32, HID / 32), 256, 0, stream>>>(Wout, WoutT);
    out_gemm<<<dim3(VOCAB / 64, (TSEQ * BATCH / 16) / 8), 256, 0, stream>>>(Hb, WoutT, bout, out);
}
